// MultiHeadAttention_15006615733773
// MI455X (gfx1250) — matrix operand order repaired and checked by reading, not compiled, not run on hardware
//
#include <hip/hip_runtime.h>
#include <hip/hip_bf16.h>
#include <math.h>

// ---------------------------------------------------------------------------
// MHA: B=2, S=2048, E=1024, H=16, D=64.  All GEMMs + attention via
// v_wmma_f32_16x16x32_bf16 (bf16 in, f32 accumulate).  Projection GEMMs
// stage the shared W tile through LDS with GLOBAL_LOAD_ASYNC_TO_LDS_B128
// (double-buffered, ASYNCcnt-tracked).
// ---------------------------------------------------------------------------

#define kB 2
#define kS 2048
#define kE 1024
#define kH 16
#define kD 64
#define kM (kB * kS)          // 4096 rows in all projection GEMMs

typedef __bf16 bf16_t;
typedef __attribute__((ext_vector_type(16))) __bf16 v16bf;
typedef __attribute__((ext_vector_type(8)))  __bf16 v8bf;
typedef __attribute__((ext_vector_type(8)))  float  v8f;

static __device__ inline v8f wmma_bf16(v16bf a, v16bf b, v8f c) {
  // (neg_a, A, neg_b, B, c_mod, C, reuse_a, reuse_b)
  return __builtin_amdgcn_wmma_f32_16x16x32_bf16(false, a, false, b,
                                                 (short)0, c, false, false);
}

// Build a 16-half fragment from two aligned 8-half (16B) loads.
static __device__ inline v16bf load_frag2(const bf16_t* p0, const bf16_t* p1) {
  v8bf a = *(const v8bf*)p0;
  v8bf b = *(const v8bf*)p1;
  v16bf r;
#pragma unroll
  for (int i = 0; i < 8; ++i) { r[i] = a[i]; r[8 + i] = b[i]; }
  return r;
}

// A-matrix 16x32 bf16 fragment: lane r holds row M=r.
// halves 0..7  -> K = hi*8 + 0..7
// halves 8..15 -> K = 16 + hi*8 + 0..7        (per ISA 16-bit A layout)
static __device__ inline v16bf load_a_frag(const bf16_t* row, int hi) {
  return load_frag2(row + hi * 8, row + 16 + hi * 8);
}

// B-matrix 32x16 bf16 fragment: lane r holds column N=r.
// halves 0..7  -> K = hi*8 + 0..7, halves 8..15 -> K = 16 + hi*8 + 0..7
// (same K order as the A fragment; B[.,r] = row r of the K-major source).
static __device__ inline v16bf load_b_frag(const bf16_t* row, int hi) {
  return load_frag2(row + hi * 8, row + 16 + hi * 8);
}

// ---------------------------------------------------------------------------
// fp32 -> bf16 cast (grid-stride)
// ---------------------------------------------------------------------------
__global__ void cast_f32_bf16(const float* __restrict__ src,
                              bf16_t* __restrict__ dst, int n) {
  int i = blockIdx.x * blockDim.x + threadIdx.x;
  int stride = gridDim.x * blockDim.x;
  for (; i < n; i += stride) dst[i] = (bf16_t)src[i];
}

// ---------------------------------------------------------------------------
// C[M,N] = A[M,K] * W[N,K]^T + bias[N]   (K = N = kE = 1024)
// Workgroup = 4 waves sharing one 64-column group (ntg); each wave owns one
// 16-row M tile.  The 64x32 W tile for the K-step is staged into LDS once per
// block via async LDS loads, double-buffered:
//   issue tile k+1 -> s_wait_asynccnt 2 (tile k landed, k+1 in flight)
//   -> barrier -> 4 WMMAs per wave from LDS B-fragments.
// MODE 0: out bf16, layout [B,H,S,D]     (Q, K heads)
// MODE 1: out bf16, layout [B,H,D,S]     (V transposed for the PV GEMM)
// MODE 2: out fp32, layout [M,N] flat    (final projection -> d_out)
// ---------------------------------------------------------------------------
#define WT_ROW 40                         // padded row stride (halves): 20 dw,
                                          // 16 rows hit 16 distinct banks
#define WT_BUF (64 * WT_ROW)              // halves per buffer
template <int MODE>
__global__ void __launch_bounds__(128)
gemm_xwT(const bf16_t* __restrict__ A, const bf16_t* __restrict__ W,
         const float* __restrict__ bias, void* __restrict__ out) {
  __shared__ __align__(16) bf16_t wtile[2 * WT_BUF];   // 10,240 B
  const int tid  = threadIdx.x;
  const int lane = tid & 31;
  const int wv   = tid >> 5;
  const int ntg  = blockIdx.x & 15;             // N/64 group (shared by block)
  const int mt   = (blockIdx.x >> 4) * 4 + wv;  // per-wave M tile
  const int r    = lane & 15;
  const int hi   = lane >> 4;

  // Stage 64 rows x 32 halves of W (rows ntg*64.. , cols kk..kk+31) into LDS
  // buffer `buf`.  256 16-byte chunks, 2 per thread, via async-to-LDS copies.
  auto stage = [&](int buf, int kk) {
#pragma unroll
    for (int q = 0; q < 2; ++q) {
      int c    = tid + q * 128;                 // chunk id 0..255
      int row  = c >> 2;                        // 0..63
      int part = c & 3;                         // 16B chunk within the row
      uint32_t loff = (uint32_t)(buf * (WT_BUF * 2) + row * (WT_ROW * 2) +
                                 part * 16);    // LDS byte offset (base 0)
      const bf16_t* g = W + (size_t)(ntg * 64 + row) * kE + kk + part * 8;
      asm volatile("global_load_async_to_lds_b128 %0, %1, off"
                   :: "v"(loff), "v"(g) : "memory");
    }
  };

  v8f acc[4];
#pragma unroll
  for (int nt = 0; nt < 4; ++nt)
#pragma unroll
    for (int j = 0; j < 8; ++j) acc[nt][j] = 0.0f;

  const bf16_t* arow = A + (size_t)(mt * 16 + r) * kE;

  stage(0, 0);                                  // prologue fill
  for (int kk = 0; kk < kE; kk += 32) {
    const int cur = (kk >> 5) & 1;
    if (kk + 32 < kE) {                         // uniform branch
      stage(cur ^ 1, kk + 32);                  // prefetch next tile
      asm volatile("s_wait_asynccnt 0x2" ::: "memory");  // cur tile landed
    } else {
      asm volatile("s_wait_asynccnt 0x0" ::: "memory");
    }
    __syncthreads();                            // all waves' chunks visible

    v16bf af = load_a_frag(arow + kk, hi);
    const bf16_t* wb = &wtile[cur * WT_BUF];
#pragma unroll
    for (int nt = 0; nt < 4; ++nt) {
      v16bf bf = load_b_frag(wb + (nt * 16 + r) * WT_ROW, hi);
      acc[nt] = wmma_bf16(af, bf, acc[nt]);
    }
    __syncthreads();                            // done reading cur buffer
  }

#pragma unroll
  for (int nt = 0; nt < 4; ++nt) {
#pragma unroll
    for (int j = 0; j < 8; ++j) {
      int m = mt * 16 + j + hi * 8;             // C layout: vgpr j, half hi
      int n = ntg * 64 + nt * 16 + r;
      float v = acc[nt][j] + bias[n];
      if (MODE == 2) {
        ((float*)out)[(size_t)m * kE + n] = v;
      } else {
        int b = m >> 11, s = m & (kS - 1);
        int h = n >> 6,  d = n & (kD - 1);
        if (MODE == 0)
          ((bf16_t*)out)[(((size_t)b * kH + h) * kS + s) * kD + d] = (bf16_t)v;
        else
          ((bf16_t*)out)[(((size_t)b * kH + h) * kD + d) * kS + s] = (bf16_t)v;
      }
    }
  }
}

// ---------------------------------------------------------------------------
// Flash attention: one wave per (b, h, 16-query tile); online softmax over
// 32-wide KV tiles.  qh/kh: [B,H,S,D] bf16, vt: [B,H,D,S] bf16,
// ao: [B,S,E] bf16.
// ---------------------------------------------------------------------------
__global__ void __launch_bounds__(128)
attn_kernel(const bf16_t* __restrict__ qh, const bf16_t* __restrict__ kh,
            const bf16_t* __restrict__ vt, bf16_t* __restrict__ ao) {
  __shared__ __align__(16) bf16_t pbuf[4][16][40];  // padded: no bank conflicts
  const int lane = threadIdx.x & 31;
  const int w    = threadIdx.x >> 5;
  const int wid  = blockIdx.x * 4 + w;          // 4096 waves = B*H*(S/16)
  const int qt   = wid & 127;                   // S/16
  const int bh   = wid >> 7;                    // b*H + h
  const int r    = lane & 15;
  const int hi   = lane >> 4;
  const float scale = 0.125f;                   // 1/sqrt(64)

  const bf16_t* Qb = qh + ((size_t)bh * kS + qt * 16) * kD;
  const bf16_t* Kb = kh + (size_t)bh * kS * kD;
  const bf16_t* Vb = vt + (size_t)bh * kD * kS;

  // Q fragments for the whole tile (d = 0..31 and 32..63), kept in registers.
  const v16bf aq0 = load_a_frag(Qb + r * kD + 0, hi);
  const v16bf aq1 = load_a_frag(Qb + r * kD + 32, hi);

  v8f oacc[4];
  float mrow[8], lrow[8];
#pragma unroll
  for (int dt = 0; dt < 4; ++dt)
#pragma unroll
    for (int j = 0; j < 8; ++j) oacc[dt][j] = 0.0f;
#pragma unroll
  for (int j = 0; j < 8; ++j) { mrow[j] = -1e30f; lrow[j] = 0.0f; }

  for (int kv0 = 0; kv0 < kS; kv0 += 32) {
    // ---- S = Q * K^T for two 16-wide KV tiles ----
    const bf16_t* k0 = Kb + (size_t)(kv0 + r) * kD;
    const bf16_t* k1 = Kb + (size_t)(kv0 + 16 + r) * kD;
    v8f s0, s1;
#pragma unroll
    for (int j = 0; j < 8; ++j) { s0[j] = 0.0f; s1[j] = 0.0f; }
    s0 = wmma_bf16(aq0, load_b_frag(k0, hi), s0);
    s0 = wmma_bf16(aq1, load_b_frag(k0 + 32, hi), s0);
    s1 = wmma_bf16(aq0, load_b_frag(k1, hi), s1);
    s1 = wmma_bf16(aq1, load_b_frag(k1 + 32, hi), s1);

    // ---- online softmax (rows live across the 16 lanes of each half) ----
#pragma unroll
    for (int j = 0; j < 8; ++j) {
      float x0 = s0[j] * scale, x1 = s1[j] * scale;
      float mx = fmaxf(x0, x1);
#pragma unroll
      for (int msk = 8; msk >= 1; msk >>= 1)
        mx = fmaxf(mx, __shfl_xor(mx, msk, 32));
      float mnew = fmaxf(mrow[j], mx);
      float corr = __expf(mrow[j] - mnew);
      mrow[j] = mnew;
      float p0 = __expf(x0 - mnew);
      float p1 = __expf(x1 - mnew);
      float ps = p0 + p1;
#pragma unroll
      for (int msk = 8; msk >= 1; msk >>= 1) ps += __shfl_xor(ps, msk, 32);
      lrow[j] = lrow[j] * corr + ps;
#pragma unroll
      for (int dt = 0; dt < 4; ++dt) oacc[dt][j] *= corr;
      int row = j + hi * 8;
      pbuf[w][row][r]      = (bf16_t)p0;        // C layout -> LDS (row, kv col)
      pbuf[w][row][16 + r] = (bf16_t)p1;
    }
    asm volatile("s_wait_dscnt 0x0" ::: "memory"); // LDS in-order; belt & braces

    // ---- re-read P as a 16x32 A fragment ----
    v16bf ap = load_a_frag(&pbuf[w][r][0], hi);

    // ---- O += P * V  (V^T rows are contiguous in kv) ----
#pragma unroll
    for (int dt = 0; dt < 4; ++dt) {
      const bf16_t* vp = Vb + (size_t)(dt * 16 + r) * kS + kv0;
      oacc[dt] = wmma_bf16(ap, load_b_frag(vp, hi), oacc[dt]);
    }
  }

  // ---- normalize and write [B,S,E] ----
  const int b = bh >> 4, h = bh & 15;
#pragma unroll
  for (int j = 0; j < 8; ++j) {
    int row = j + hi * 8;
    int s = qt * 16 + row;
    float inv = 1.0f / lrow[j];
#pragma unroll
    for (int dt = 0; dt < 4; ++dt) {
      int e = h * kD + dt * 16 + r;
      ao[((size_t)b * kS + s) * kE + e] = (bf16_t)(oacc[dt][j] * inv);
    }
  }
}

// ---------------------------------------------------------------------------
// Host-side orchestration
// ---------------------------------------------------------------------------
extern "C" void kernel_launch(void* const* d_in, const int* in_sizes, int n_in,
                              void* d_out, int out_size, void* d_ws,
                              size_t ws_size, hipStream_t stream) {
  const float* query = (const float*)d_in[0];
  const float* key   = (const float*)d_in[1];
  const float* value = (const float*)d_in[2];
  const float* Wq = (const float*)d_in[3];  const float* bq = (const float*)d_in[4];
  const float* Wk = (const float*)d_in[5];  const float* bk = (const float*)d_in[6];
  const float* Wv = (const float*)d_in[7];  const float* bv = (const float*)d_in[8];
  const float* Wo = (const float*)d_in[9];  const float* bo = (const float*)d_in[10];

  const size_t SZ_X = (size_t)kM * kE;   // 4,194,304 elems
  const size_t SZ_W = (size_t)kE * kE;   // 1,048,576 elems

  bf16_t* p   = (bf16_t*)d_ws;           // total: 32M bf16 = 64 MB
  bf16_t* qx  = p; p += SZ_X;
  bf16_t* kx  = p; p += SZ_X;
  bf16_t* vx  = p; p += SZ_X;
  bf16_t* wqb = p; p += SZ_W;
  bf16_t* wkb = p; p += SZ_W;
  bf16_t* wvb = p; p += SZ_W;
  bf16_t* wob = p; p += SZ_W;
  bf16_t* qhh = p; p += SZ_X;            // [B,H,S,D]
  bf16_t* khh = p; p += SZ_X;            // [B,H,S,D]
  bf16_t* vtt = p; p += SZ_X;            // [B,H,D,S]
  bf16_t* ao  = p; p += SZ_X;            // [B,S,E]

  cast_f32_bf16<<<2048, 256, 0, stream>>>(query, qx, (int)SZ_X);
  cast_f32_bf16<<<2048, 256, 0, stream>>>(key,   kx, (int)SZ_X);
  cast_f32_bf16<<<2048, 256, 0, stream>>>(value, vx, (int)SZ_X);
  cast_f32_bf16<<<1024, 256, 0, stream>>>(Wq, wqb, (int)SZ_W);
  cast_f32_bf16<<<1024, 256, 0, stream>>>(Wk, wkb, (int)SZ_W);
  cast_f32_bf16<<<1024, 256, 0, stream>>>(Wv, wvb, (int)SZ_W);
  cast_f32_bf16<<<1024, 256, 0, stream>>>(Wo, wob, (int)SZ_W);

  // Projections: 4096 waves each (1024 blocks x 4 waves)
  gemm_xwT<0><<<1024, 128, 0, stream>>>(qx, wqb, bq, qhh);
  gemm_xwT<0><<<1024, 128, 0, stream>>>(kx, wkb, bk, khh);
  gemm_xwT<1><<<1024, 128, 0, stream>>>(vx, wvb, bv, vtt);

  // Attention: 4096 waves (one per (b,h,q-tile))
  attn_kernel<<<1024, 128, 0, stream>>>(qhh, khh, vtt, ao);

  // Output projection -> fp32 d_out
  gemm_xwT<2><<<1024, 128, 0, stream>>>(ao, wob, bo, d_out);
}
